// SLSTM_30674656428069
// MI455X (gfx1250) — compile-verified, matrix operand order checked by reading, not run on hardware
//
#include <hip/hip_runtime.h>
#include <cstdint>

// ---------------------------------------------------------------------------
// sLSTM message-passing forward for MI455X (gfx1250, wave32, WMMA).
// Dominant cost: fs = ihs @ W_n  (16384 x 2560 x 3584) x 9 iterations,
// executed as bf16 WMMA (v_wmma_f32_16x16x32_bf16) with fp32 accumulation.
// GEMM uses double-buffered LDS tiles; the A tile is staged with the CDNA5
// async copy (global_load_async_to_lds_b128, ASYNCcnt) and the B tile is
// register-prefetched so global latency overlaps the WMMA burst.
// ---------------------------------------------------------------------------

typedef __attribute__((ext_vector_type(16))) __bf16 v16bf;
typedef __attribute__((ext_vector_type(8)))  float  v8f;
typedef __attribute__((ext_vector_type(4)))  int    v4i;

static constexpr int kB   = 32;
static constexpr int kL   = 512;
static constexpr int kE   = 300;
static constexpr int kH   = 512;
static constexpr int kLp1 = kL + 1;

__device__ __forceinline__ unsigned short f2bf(float f) {
  unsigned int u = __float_as_uint(f);
  u += 0x7FFFu + ((u >> 16) & 1u);          // round-to-nearest-even
  return (unsigned short)(u >> 16);
}
__device__ __forceinline__ float sig_(float x) { return 1.f / (1.f + __expf(-x)); }

// --- CDNA5 async global->LDS copy (ASYNCcnt), with safe fallback ----------
#if __has_builtin(__builtin_amdgcn_global_load_async_to_lds_b128)
#define HAS_ASYNC_LDS 1
typedef __attribute__((address_space(1))) v4i* gv4i_p;   // global 128b chunk
typedef __attribute__((address_space(3))) v4i* lv4i_p;   // LDS 128b chunk

__device__ __forceinline__ gv4i_p as_global_v4i(const void* p) {
  return (gv4i_p)(unsigned long long)(uintptr_t)p;
}
__device__ __forceinline__ lv4i_p as_lds_v4i(void* p) {
  // flat LDS address: low 32 bits are the LDS byte offset (ISA 10.2 aperture)
  return (lv4i_p)(unsigned)(uintptr_t)p;
}
#else
#define HAS_ASYNC_LDS 0
#endif

__device__ __forceinline__ void wait_async0() {
#if __has_builtin(__builtin_amdgcn_s_wait_asynccnt)
  __builtin_amdgcn_s_wait_asynccnt(0);
#elif HAS_ASYNC_LDS
  asm volatile("s_wait_asynccnt 0x0" ::: "memory");
#endif
}

// ---------------------------------------------------------------------------
// Generic bf16 GEMM:  C[M,N] = A[M,K] * B[K,N] (+ bias[N]), fp32 out.
// Block tile 128x128, BK=32, 8 waves (2x4), wave tile 64x32 (4x2 WMMA accs).
// Fragment layout per ISA 7.12.2:
//   element e (0..15) of lane (idx + 16*((k%16)/8)) holds k = 16*(e/8)+8*half+(e%8)
// ---------------------------------------------------------------------------
#define BM 128
#define BN 128
#define BK 32
#define LDSK 40   // padded LDS row length (elements) to spread banks

union Frag16 { v16bf v; uint4 u[2]; };
union BStage { uint4 q[2]; unsigned short s[16]; };

__global__ __launch_bounds__(256)
void gemm_bf16_wmma(const unsigned short* __restrict__ A,
                    const unsigned short* __restrict__ Bm,
                    const float* __restrict__ bias,
                    float* __restrict__ C,
                    int M, int N, int K)
{
  __shared__ unsigned short As[2][BM * LDSK];
  __shared__ unsigned short Bs[2][BN * LDSK];

  const int tid  = threadIdx.x;
  const int lane = tid & 31;
  const int wid  = tid >> 5;
  const int half = lane >> 4;
  const int l16  = lane & 15;

  const int m0 = blockIdx.y * BM;
  const int n0 = blockIdx.x * BN;

  const int wm0 = (wid >> 2) * 64;   // 0 or 64
  const int wn0 = (wid & 3)  * 32;   // 0,32,64,96

  const bool fastA = ((K & 7) == 0);
  const bool fastB = ((N & 7) == 0);

  // ---- A tile staging: BM x BK, row-major, 16B per task, 2 tasks/thread.
  // In-range full chunks go through the CDNA5 async copy straight to LDS;
  // edge chunks take a zero-filled synchronous path.
  auto stage_a = [&](int buf, int k0) {
#pragma unroll
    for (int t = 0; t < 2; t++) {
      const int ta = tid + 256 * t;
      const int r  = ta >> 2;
      const int c  = (ta & 3) * 8;
      const int gm = m0 + r, gk = k0 + c;
      unsigned short* dst = &As[buf][r * LDSK + c];
      const unsigned short* src = A + (size_t)gm * K + gk;
      const bool inr = (gm < M) && (gk + 7 < K);
#if HAS_ASYNC_LDS
      if (fastA && inr) {
        __builtin_amdgcn_global_load_async_to_lds_b128(
            as_global_v4i(src), as_lds_v4i(dst), 0, 0);
        continue;
      }
#endif
      union { uint4 q; unsigned short s[8]; } tmp;
      if (fastA && inr) {
        tmp.q = *(const uint4*)src;
      } else {
#pragma unroll
        for (int i = 0; i < 8; i++)
          tmp.s[i] = (gm < M && gk + i < K) ? src[i] : (unsigned short)0;
      }
      *(uint4*)dst = tmp.q;
    }
  };

  // ---- B tile: BK x BN read coalesced into regs (prefetch), committed to
  // LDS transposed (Bs[n][k]) after the WMMA burst.
  auto stage_b_issue = [&](int k0, BStage& sv) {
    const int r = tid >> 3;           // k row 0..31
    const int c = (tid & 7) * 16;     // n col
    const int gk = k0 + r;
    if (fastB && gk < K && n0 + c + 15 < N) {
      sv.q[0] = *(const uint4*)(Bm + (size_t)gk * N + n0 + c);
      sv.q[1] = *(const uint4*)(Bm + (size_t)gk * N + n0 + c + 8);
    } else {
#pragma unroll
      for (int i = 0; i < 16; i++)
        sv.s[i] = (gk < K && n0 + c + i < N) ? Bm[(size_t)gk * N + n0 + c + i]
                                             : (unsigned short)0;
    }
  };
  auto stage_b_commit = [&](int buf, const BStage& sv) {
    const int r = tid >> 3;
    const int c = (tid & 7) * 16;
#pragma unroll
    for (int i = 0; i < 16; i++) Bs[buf][(c + i) * LDSK + r] = sv.s[i];
  };

  v8f acc[4][2];
#pragma unroll
  for (int i = 0; i < 4; i++)
#pragma unroll
    for (int j = 0; j < 2; j++)
#pragma unroll
      for (int e = 0; e < 8; e++) acc[i][j][e] = 0.f;

  // ---- prologue: stage tile 0 into buffer 0 ----
  {
    stage_a(0, 0);
    BStage b0;
    stage_b_issue(0, b0);
    stage_b_commit(0, b0);
    wait_async0();
    __syncthreads();
  }

  int buf = 0;
  for (int k0 = 0; k0 < K; k0 += BK) {
    const bool haveNext = (k0 + BK) < K;
    BStage nb;
    if (haveNext) {
      stage_a(buf ^ 1, k0 + BK);     // async copy into the other buffer
      stage_b_issue(k0 + BK, nb);    // global -> regs, overlaps WMMAs below
    }

    // ---- fragments + WMMA from current buffer ----
    Frag16 fa[4], fb[2];
#pragma unroll
    for (int mi = 0; mi < 4; mi++) {
      const unsigned short* p = &As[buf][(wm0 + 16 * mi + l16) * LDSK];
      fa[mi].u[0] = *(const uint4*)(p + 8 * half);
      fa[mi].u[1] = *(const uint4*)(p + 16 + 8 * half);
    }
#pragma unroll
    for (int ni = 0; ni < 2; ni++) {
      const unsigned short* p = &Bs[buf][(wn0 + 16 * ni + l16) * LDSK];
      fb[ni].u[0] = *(const uint4*)(p + 8 * half);
      fb[ni].u[1] = *(const uint4*)(p + 16 + 8 * half);
    }
#pragma unroll
    for (int mi = 0; mi < 4; mi++)
#pragma unroll
      for (int ni = 0; ni < 2; ni++)
        acc[mi][ni] = __builtin_amdgcn_wmma_f32_16x16x32_bf16(
            false, fa[mi].v, false, fb[ni].v, (short)0, acc[mi][ni],
            false, false);

    if (haveNext) stage_b_commit(buf ^ 1, nb);
    wait_async0();        // our async transfers into buf^1 have landed
    __syncthreads();      // everyone's stores/copies visible
    buf ^= 1;
  }

  // ---- epilogue: bias + store ----
#pragma unroll
  for (int mi = 0; mi < 4; mi++)
#pragma unroll
    for (int ni = 0; ni < 2; ni++) {
      const int n = n0 + wn0 + 16 * ni + l16;
      if (n < N) {
        const float bv = bias ? bias[n] : 0.f;
#pragma unroll
        for (int j = 0; j < 8; j++) {
          const int m = m0 + wm0 + 16 * mi + j + 8 * half;
          if (m < M) C[(size_t)m * N + n] = acc[mi][ni][j] + bv;
        }
      }
    }
}

// ---------------------------------------------------------------------------
// Elementwise / small kernels
// ---------------------------------------------------------------------------
__global__ void cvt_f32_bf16(const float* __restrict__ in,
                             unsigned short* __restrict__ out, long n) {
  long i = (long)blockIdx.x * blockDim.x + threadIdx.x;
  if (i < n) out[i] = f2bf(in[i]);
}

__global__ void msum_kernel(const float* __restrict__ mask, float* __restrict__ msum) {
  int b = threadIdx.x;
  if (b < kB) {
    float s = 0.f;
    for (int l = 0; l < kL; l++) s += mask[b * kL + l];
    msum[b] = s;
  }
}

// nhs = ncs = embs
__global__ void copy_state(const float* __restrict__ embs,
                           float* __restrict__ nhs, float* __restrict__ ncs, long n) {
  long i = (long)blockIdx.x * blockDim.x + threadIdx.x;
  if (i < n) { float v = embs[i]; nhs[i] = v; ncs[i] = v; }
}

// gh = gc = embs.sum(L) / msum      (thread per (b,h))
__global__ void init_g_kernel(const float* __restrict__ embs, const float* __restrict__ msum,
                              float* __restrict__ gh, float* __restrict__ gc) {
  int i = blockIdx.x * blockDim.x + threadIdx.x;
  if (i >= kB * kH) return;
  int b = i / kH, h = i % kH;
  float s = 0.f;
  for (int l = 0; l < kL; l++) s += embs[((size_t)b * kL + l) * kH + h];
  float v = s / msum[b];
  gh[i] = v; gc[i] = v;
}

// h_bar = nhs.sum(L) / msum
__global__ void hbar_kernel(const float* __restrict__ nhs, const float* __restrict__ msum,
                            float* __restrict__ hbar) {
  int i = blockIdx.x * blockDim.x + threadIdx.x;
  if (i >= kB * kH) return;
  int b = i / kH, h = i % kH;
  float s = 0.f;
  for (int l = 0; l < kL; l++) s += nhs[((size_t)b * kL + l) * kH + h];
  hbar[i] = s / msum[b];
}

// xg[b,l,0:H]=gh ; xg[b,l,H:2H]= l==0 ? h_bar : nhs[l-1]   (bf16)
__global__ void build_xg(const float* __restrict__ gh, const float* __restrict__ hbar,
                         const float* __restrict__ nhs, unsigned short* __restrict__ xg) {
  int i = blockIdx.x * blockDim.x + threadIdx.x;
  if (i >= kB * kLp1 * kH) return;
  int b = i / (kLp1 * kH);
  int r = i % (kLp1 * kH);
  int l = r / kH, h = r % kH;
  size_t row = (size_t)b * kLp1 + l;
  xg[row * (2 * kH) + h] = f2bf(gh[b * kH + h]);
  float v = (l == 0) ? hbar[b * kH + h] : nhs[((size_t)b * kL + (l - 1)) * kH + h];
  xg[row * (2 * kH) + kH + h] = f2bf(v);
}

// softmax over L+1 of fs2 (bias already fused), weighted by ics -> n_gc
__global__ void g_softmax_kernel(const float* __restrict__ fs2, const float* __restrict__ mask,
                                 const float* __restrict__ gc, const float* __restrict__ ncs,
                                 float* __restrict__ ngc) {
  int i = blockIdx.x * blockDim.x + threadIdx.x;
  if (i >= kB * kH) return;
  int b = i / kH, h = i % kH;
  float mx = -1e30f;
  for (int l = 0; l < kLp1; l++) {
    float pen = (l < kL) ? (1.f - mask[b * kL + l]) * 200.f : 0.f;
    float v = fs2[((size_t)b * kLp1 + l) * kH + h] + pen;
    mx = fmaxf(mx, v);
  }
  float se = 0.f, ws = 0.f;
  for (int l = 0; l < kLp1; l++) {
    float pen = (l < kL) ? (1.f - mask[b * kL + l]) * 200.f : 0.f;
    float e = __expf(fs2[((size_t)b * kLp1 + l) * kH + h] + pen - mx);
    float ic = (l == 0) ? gc[b * kH + h] : ncs[((size_t)b * kL + (l - 1)) * kH + h];
    se += e; ws += e * ic;
  }
  ngc[i] = ws / se;
}

__global__ void build_xo(const float* __restrict__ gh, const float* __restrict__ hbar,
                         unsigned short* __restrict__ xo) {
  int i = blockIdx.x * blockDim.x + threadIdx.x;
  if (i >= kB * kH) return;
  int b = i / kH, h = i % kH;
  xo[(size_t)b * 2 * kH + h]      = f2bf(gh[i]);
  xo[(size_t)b * 2 * kH + kH + h] = f2bf(hbar[i]);
}

// gh2 = sigmoid(gate_lin) * tanh(ngc) ; gc2 = ngc   (b_gout fused in GEMM)
__global__ void g_finalize(const float* __restrict__ gate_lin, const float* __restrict__ ngc,
                           float* __restrict__ gh2, float* __restrict__ gc2) {
  int i = blockIdx.x * blockDim.x + threadIdx.x;
  if (i >= kB * kH) return;
  gh2[i] = sig_(gate_lin[i]) * tanhf(ngc[i]);
  gc2[i] = ngc[i];
}

// ihs[b,l,:] = [nhs[l-1], nhs[l], nhs[l+1], embs, gh]   (bf16)
__global__ void build_ihs(const float* __restrict__ nhs, const float* __restrict__ embs,
                          const float* __restrict__ gh, unsigned short* __restrict__ ihs) {
  int i = blockIdx.x * blockDim.x + threadIdx.x;
  if (i >= kB * kL * kH) return;
  int b = i / (kL * kH);
  int r = i % (kL * kH);
  int l = r / kH, h = r % kH;
  size_t row = (size_t)b * kL + l;
  unsigned short* o = ihs + row * (size_t)(5 * kH);
  o[h]            = f2bf(l > 0      ? nhs[(row - 1) * kH + h] : 0.f);
  o[kH + h]       = f2bf(nhs[row * kH + h]);
  o[2 * kH + h]   = f2bf(l < kL - 1 ? nhs[(row + 1) * kH + h] : 0.f);
  o[3 * kH + h]   = f2bf(embs[row * kH + h]);
  o[4 * kH + h]   = f2bf(gh[b * kH + h]);
}

// og=sigmoid(fs[:H]); gs=softmax5(fs[2H:7H]); n_c = sum gs*ics; apply mask
__global__ void node_post(const float* __restrict__ fs, const float* __restrict__ ncs,
                          const float* __restrict__ gc, const float* __restrict__ embs,
                          const float* __restrict__ mask,
                          float* __restrict__ nhs_o, float* __restrict__ ncs_o) {
  int i = blockIdx.x * blockDim.x + threadIdx.x;
  if (i >= kB * kL * kH) return;
  int b = i / (kL * kH);
  int r = i % (kL * kH);
  int l = r / kH, h = r % kH;
  size_t row = (size_t)b * kL + l;
  const float* fr = fs + row * (size_t)(7 * kH);
  float og = sig_(fr[h]);
  float g[5];
#pragma unroll
  for (int s = 0; s < 5; s++) g[s] = fr[2 * kH + s * kH + h];
  float mx = g[0];
#pragma unroll
  for (int s = 1; s < 5; s++) mx = fmaxf(mx, g[s]);
  float se = 0.f;
#pragma unroll
  for (int s = 0; s < 5; s++) { g[s] = __expf(g[s] - mx); se += g[s]; }
  float ic0 = (l > 0)      ? ncs[(row - 1) * kH + h] : 0.f;
  float ic1 = ncs[row * kH + h];
  float ic2 = (l < kL - 1) ? ncs[(row + 1) * kH + h] : 0.f;
  float ic3 = gc[b * kH + h];
  float ic4 = embs[row * kH + h];
  float nc = (g[0] * ic0 + g[1] * ic1 + g[2] * ic2 + g[3] * ic3 + g[4] * ic4) / se;
  float m = mask[b * kL + l];
  nhs_o[row * kH + h] = m * og * tanhf(nc);
  ncs_o[row * kH + h] = m * nc;
}

// rep = (nhs.sum(L) + gh) / msum
__global__ void rep_kernel(const float* __restrict__ nhs, const float* __restrict__ gh,
                           const float* __restrict__ msum, float* __restrict__ rep) {
  int i = blockIdx.x * blockDim.x + threadIdx.x;
  if (i >= kB * kH) return;
  int b = i / kH, h = i % kH;
  float s = gh[i];
  for (int l = 0; l < kL; l++) s += nhs[((size_t)b * kL + l) * kH + h];
  rep[i] = s / msum[b];
}

// per-batch: rep2 = tanh(rep@W_up+b_up) ; out = log_softmax(rep2@W_fc+b_fc)
__global__ __launch_bounds__(256)
void head_kernel(const float* __restrict__ rep, const float* __restrict__ W_up,
                 const float* __restrict__ b_up, const float* __restrict__ W_fc,
                 const float* __restrict__ b_fc, float* __restrict__ out) {
  int b = blockIdx.x, tid = threadIdx.x;
  __shared__ float rloc[kH];
  __shared__ float r2[2 * kH];
  for (int k = tid; k < kH; k += 256) rloc[k] = rep[b * kH + k];
  __syncthreads();
  for (int j = tid; j < 2 * kH; j += 256) {
    float s = b_up[j];
    for (int k = 0; k < kH; k++) s += rloc[k] * W_up[(size_t)k * (2 * kH) + j];
    r2[j] = tanhf(s);
  }
  __syncthreads();
  if (tid == 0) {
    float z[2];
    for (int c = 0; c < 2; c++) {
      float s = b_fc[c];
      for (int j = 0; j < 2 * kH; j++) s += r2[j] * W_fc[j * 2 + c];
      z[c] = s;
    }
    float m = fmaxf(z[0], z[1]);
    float lse = m + logf(__expf(z[0] - m) + __expf(z[1] - m));
    out[b * 2 + 0] = z[0] - lse;
    out[b * 2 + 1] = z[1] - lse;
  }
}

// ---------------------------------------------------------------------------
// Host orchestration
// ---------------------------------------------------------------------------
extern "C" void kernel_launch(void* const* d_in, const int* in_sizes, int n_in,
                              void* d_out, int out_size, void* d_ws, size_t ws_size,
                              hipStream_t stream) {
  (void)in_sizes; (void)n_in; (void)out_size; (void)ws_size;

  const float* data   = (const float*)d_in[0];
  const float* mask   = (const float*)d_in[1];
  const float* W_in   = (const float*)d_in[2];
  const float* b_in   = (const float*)d_in[3];
  const float* W_n    = (const float*)d_in[4];
  const float* b_n    = (const float*)d_in[5];
  const float* W_gatt = (const float*)d_in[6];
  const float* b_gatt = (const float*)d_in[7];
  const float* W_gout = (const float*)d_in[8];
  const float* b_gout = (const float*)d_in[9];
  const float* W_up   = (const float*)d_in[10];
  const float* b_up   = (const float*)d_in[11];
  const float* W_fc   = (const float*)d_in[12];
  const float* b_fc   = (const float*)d_in[13];
  float* out = (float*)d_out;

  char* ws = (char*)d_ws;
  size_t off = 0;
  auto alloc = [&](size_t bytes) -> void* {
    void* p = ws + off;
    off += (bytes + 255) & ~(size_t)255;
    return p;
  };

  const size_t BLH = (size_t)kB * kL * kH;

  unsigned short* data_bf  = (unsigned short*)alloc((size_t)kB * kL * kE * 2);
  unsigned short* Win_bf   = (unsigned short*)alloc((size_t)kE * kH * 2);
  unsigned short* Wn_bf    = (unsigned short*)alloc((size_t)5 * kH * 7 * kH * 2);
  unsigned short* Wgatt_bf = (unsigned short*)alloc((size_t)2 * kH * kH * 2);
  unsigned short* Wgout_bf = (unsigned short*)alloc((size_t)2 * kH * kH * 2);
  float* embs   = (float*)alloc(BLH * 4);
  float* nhsA   = (float*)alloc(BLH * 4);
  float* nhsB   = (float*)alloc(BLH * 4);
  float* ncsA   = (float*)alloc(BLH * 4);
  float* ncsB   = (float*)alloc(BLH * 4);
  float* ghA    = (float*)alloc((size_t)kB * kH * 4);
  float* ghB    = (float*)alloc((size_t)kB * kH * 4);
  float* gcA    = (float*)alloc((size_t)kB * kH * 4);
  float* gcB    = (float*)alloc((size_t)kB * kH * 4);
  float* msum   = (float*)alloc(kB * 4);
  float* hbar   = (float*)alloc((size_t)kB * kH * 4);
  unsigned short* xg = (unsigned short*)alloc((size_t)kB * kLp1 * 2 * kH * 2);
  float* fs2    = (float*)alloc((size_t)kB * kLp1 * kH * 4);
  unsigned short* xo = (unsigned short*)alloc((size_t)kB * 2 * kH * 2);
  float* gate_lin = (float*)alloc((size_t)kB * kH * 4);
  float* ngc    = (float*)alloc((size_t)kB * kH * 4);
  unsigned short* ihs = (unsigned short*)alloc(BLH * 5 * 2);
  float* fs     = (float*)alloc((size_t)kB * kL * 7 * kH * 4);
  float* rep    = (float*)alloc((size_t)kB * kH * 4);

  auto nblk = [](long n) { return (unsigned)((n + 255) / 256); };
  auto gemm = [&](const unsigned short* A, const unsigned short* Bm,
                  const float* bias, float* C, int M, int N, int K) {
    dim3 g((N + BN - 1) / BN, (M + BM - 1) / BM);
    gemm_bf16_wmma<<<g, 256, 0, stream>>>(A, Bm, bias, C, M, N, K);
  };

  // ---- weight / input down-conversion to bf16 ----
  cvt_f32_bf16<<<nblk((long)kB * kL * kE), 256, 0, stream>>>(data, data_bf, (long)kB * kL * kE);
  cvt_f32_bf16<<<nblk((long)kE * kH), 256, 0, stream>>>(W_in, Win_bf, (long)kE * kH);
  cvt_f32_bf16<<<nblk((long)5 * kH * 7 * kH), 256, 0, stream>>>(W_n, Wn_bf, (long)5 * kH * 7 * kH);
  cvt_f32_bf16<<<nblk((long)2 * kH * kH), 256, 0, stream>>>(W_gatt, Wgatt_bf, (long)2 * kH * kH);
  cvt_f32_bf16<<<nblk((long)2 * kH * kH), 256, 0, stream>>>(W_gout, Wgout_bf, (long)2 * kH * kH);

  // ---- init: embs, state, g ----
  msum_kernel<<<1, 32, 0, stream>>>(mask, msum);
  gemm(data_bf, Win_bf, b_in, embs, kB * kL, kH, kE);
  copy_state<<<nblk((long)BLH), 256, 0, stream>>>(embs, nhsA, ncsA, (long)BLH);
  init_g_kernel<<<nblk((long)kB * kH), 256, 0, stream>>>(embs, msum, ghA, gcA);

  float* nhs_c = nhsA; float* nhs_n = nhsB;
  float* ncs_c = ncsA; float* ncs_n = ncsB;
  float* gh_c  = ghA;  float* gh_n  = ghB;
  float* gc_c  = gcA;  float* gc_n  = gcB;

  for (int it = 0; it < 9; it++) {
    // ---- update_g (uses old state) ----
    hbar_kernel<<<nblk((long)kB * kH), 256, 0, stream>>>(nhs_c, msum, hbar);
    build_xg<<<nblk((long)kB * kLp1 * kH), 256, 0, stream>>>(gh_c, hbar, nhs_c, xg);
    gemm(xg, Wgatt_bf, b_gatt, fs2, kB * kLp1, kH, 2 * kH);
    g_softmax_kernel<<<nblk((long)kB * kH), 256, 0, stream>>>(fs2, mask, gc_c, ncs_c, ngc);
    build_xo<<<nblk((long)kB * kH), 256, 0, stream>>>(gh_c, hbar, xo);
    gemm(xo, Wgout_bf, b_gout, gate_lin, kB, kH, 2 * kH);

    // ---- update_nodes (uses old gh/gc) ----
    build_ihs<<<nblk((long)BLH), 256, 0, stream>>>(nhs_c, embs, gh_c, ihs);
    gemm(ihs, Wn_bf, b_n, fs, kB * kL, 7 * kH, 5 * kH);
    node_post<<<nblk((long)BLH), 256, 0, stream>>>(fs, ncs_c, gc_c, embs, mask, nhs_n, ncs_n);
    g_finalize<<<nblk((long)kB * kH), 256, 0, stream>>>(gate_lin, ngc, gh_n, gc_n);

    // ---- swap double buffers ----
    float* t;
    t = nhs_c; nhs_c = nhs_n; nhs_n = t;
    t = ncs_c; ncs_c = ncs_n; ncs_n = t;
    t = gh_c;  gh_c  = gh_n;  gh_n  = t;
    t = gc_c;  gc_c  = gc_n;  gc_n  = t;
  }

  // ---- head ----
  rep_kernel<<<nblk((long)kB * kH), 256, 0, stream>>>(nhs_c, gh_c, msum, rep);
  head_kernel<<<kB, 256, 0, stream>>>(rep, W_up, b_up, W_fc, b_fc, out);
}
